// GPTAttention_67834713473770
// MI455X (gfx1250) — matrix operand order repaired and checked by reading, not compiled, not run on hardware
//
#include <hip/hip_runtime.h>
#include <hip/hip_bf16.h>

// ---------------------------------------------------------------------------
// GPT attention block for gfx1250 (MI455X), wave32, WMMA bf16 path with
// async-to-LDS double-buffered GEMM staging.
//   pass 0: x -> bf16; W_qkv, W_out -> transposed bf16 (one-time, ~1us traffic)
//   pass 1: QKV GEMM (async LDS staging) -> Q,K,V head-major bf16 (Q * 1/8)
//   pass 2: causal flash attention (WMMA QK^T + PV, online softmax)
//   pass 3: output GEMM -> f32 + bias
// B=4, S=2048, D=1024, H=16, d=64
// ---------------------------------------------------------------------------

typedef __bf16 bf16_t;
typedef __attribute__((ext_vector_type(16))) __bf16 v16bf;
typedef __attribute__((ext_vector_type(8)))  __bf16 v8bf;
typedef __attribute__((ext_vector_type(8)))  float  v8f;

#define BATCH 4
#define SEQ   2048
#define DMODEL 1024
#define NHEADS 16
#define HDIM  64

__device__ __forceinline__ v8f wmma_bf16(v16bf a, v16bf b, v8f c) {
  return __builtin_amdgcn_wmma_f32_16x16x32_bf16(false, a, false, b,
                                                 (short)0, c, false, false);
}

// ---- async global->LDS copy (16B per lane), CDNA5 ASYNCcnt path -----------
#if __has_builtin(__builtin_amdgcn_global_load_async_to_lds_b128)
#define HAS_ASYNC_LDS 1
#else
#define HAS_ASYNC_LDS 0
#endif

typedef int v4i_gcc __attribute__((vector_size(4 * sizeof(int))));
typedef __attribute__((address_space(1))) v4i_gcc gv4i;   // global int4
typedef __attribute__((address_space(3))) v4i_gcc lv4i;   // LDS int4

__device__ __forceinline__ void async_copy16(void* lds, const void* g) {
#if HAS_ASYNC_LDS
  __builtin_amdgcn_global_load_async_to_lds_b128(
      (gv4i*)(void*)g, (lv4i*)(void*)lds, 0, 0);
#else
  *(v8bf*)lds = *(const v8bf*)g;   // synchronous fallback
#endif
}

template <int N>
__device__ __forceinline__ void async_wait() {
#if HAS_ASYNC_LDS
#if __has_builtin(__builtin_amdgcn_s_wait_asynccnt)
  __builtin_amdgcn_s_wait_asynccnt(N);
#else
  asm volatile("s_wait_asynccnt %0" ::"i"(N) : "memory");
#endif
#endif
}

// A-matrix fragment (16x32 bf16, MxK): lane = M (lane%16), half = lane/16.
// Per lane 16 elems: K runs [half*8 .. +7] and [16+half*8 .. +7].
__device__ __forceinline__ v16bf frag_ld_a(const bf16_t* row, int colbase, int half) {
  union { v16bf v; v8bf h[2]; } u;
  u.h[0] = *(const v8bf*)(row + colbase + half * 8);
  u.h[1] = *(const v8bf*)(row + colbase + 16 + half * 8);
  return u.v;
}

// B-matrix fragment (32x16 bf16, KxN): lane = N (lane%16), K runs [half*8 .. +7] and [16+half*8 .. +7].
// `row` points at the N-major row (tile stored transposed: Bt[n][k]).
__device__ __forceinline__ v16bf frag_ld_b(const bf16_t* row, int colbase, int half) {
  union { v16bf v; v8bf h[2]; } u;
  u.h[0] = *(const v8bf*)(row + colbase + half * 8);
  u.h[1] = *(const v8bf*)(row + colbase + 16 + half * 8);
  return u.v;
}

// ---------------------------------------------------------------------------
// One-time conversions
// ---------------------------------------------------------------------------
__global__ __launch_bounds__(256) void convert_f32_bf16(
    const float* __restrict__ src, bf16_t* __restrict__ dst) {
  size_t idx = (size_t)blockIdx.x * 256 + threadIdx.x;   // float4 index
  float4 f = *(const float4*)(src + idx * 4);
  bf16_t* d = dst + idx * 4;
  d[0] = (bf16_t)f.x; d[1] = (bf16_t)f.y;
  d[2] = (bf16_t)f.z; d[3] = (bf16_t)f.w;
}

// W[K][N] f32 -> Wt[N][K] bf16, 32x32 LDS tiles
__global__ __launch_bounds__(256) void transpose_to_bf16(
    const float* __restrict__ W, bf16_t* __restrict__ Wt, int K, int N) {
  __shared__ float tile[32][33];
  const int n0 = blockIdx.x * 32, k0 = blockIdx.y * 32;
  const int tid = threadIdx.x;
#pragma unroll
  for (int it = 0; it < 4; ++it) {
    int idx = tid + 256 * it;
    int r = idx >> 5, c = idx & 31;
    tile[r][c] = W[(size_t)(k0 + r) * N + n0 + c];
  }
  __syncthreads();
#pragma unroll
  for (int it = 0; it < 4; ++it) {
    int idx = tid + 256 * it;
    int r = idx >> 5, c = idx & 31;        // r = n-local, c = k-local
    Wt[(size_t)(n0 + r) * K + k0 + c] = (bf16_t)tile[c][r];
  }
}

// ---------------------------------------------------------------------------
// GEMM: C[M,N] = A[M,K] * Wt[N,K]^T + bias.  A, Wt are bf16.
// 128x128 block tile, 8 waves (4x2), K-step 32, double-buffered async staging.
// QKV: scatter epilogue to Q/K/V head-major bf16 (Q scaled 1/8);
// else f32 row-major + bias to Cout.
// ---------------------------------------------------------------------------
template <bool QKV>
__global__ __launch_bounds__(256) void gemm128(
    const bf16_t* __restrict__ A, const bf16_t* __restrict__ Wt,
    const float* __restrict__ bias, float* __restrict__ Cout,
    bf16_t* __restrict__ Qb, bf16_t* __restrict__ Kb, bf16_t* __restrict__ Vb,
    int M, int N, int K) {
  constexpr int LDT = 40;  // bf16 per LDS tile row (32 + 8 pad), 16B-aligned
  __shared__ __align__(16) bf16_t As[2][128 * LDT];   // A tile [m][k]
  __shared__ __align__(16) bf16_t Bs[2][128 * LDT];   // Wt tile [n][k]

  const int tid = threadIdx.x;
  const int wave = tid >> 5, lane = tid & 31;
  const int half = lane >> 4, l16 = lane & 15;
  const int wm = wave & 3, wn = wave >> 2;            // 4 (M) x 2 (N)
  const int rowBlk = blockIdx.y * 128;
  const int colBlk = blockIdx.x * 128;

  // stage one 128x32 bf16 tile pair: 2+2 async b128 per thread (4/wave)
  auto stage = [&](int k0, int buf) {
#pragma unroll
    for (int it = 0; it < 2; ++it) {
      int idx = tid + 256 * it;                        // 512 x 16B chunks
      int r = idx >> 2, c8 = idx & 3;
      async_copy16(&As[buf][r * LDT + c8 * 8],
                   A + (size_t)(rowBlk + r) * K + k0 + c8 * 8);
    }
#pragma unroll
    for (int it = 0; it < 2; ++it) {
      int idx = tid + 256 * it;
      int n = idx >> 2, c8 = idx & 3;
      async_copy16(&Bs[buf][n * LDT + c8 * 8],
                   Wt + (size_t)(colBlk + n) * K + k0 + c8 * 8);
    }
  };

  v8f acc[2][4] = {};
  stage(0, 0);
  for (int k0 = 0; k0 < K; k0 += 32) {
    const int buf = (k0 >> 5) & 1;
    if (k0 + 32 < K) {
      stage(k0 + 32, buf ^ 1);
      async_wait<4>();          // current tile resident; next tile in flight
    } else {
      async_wait<0>();
    }
    __syncthreads();

    v16bf afrag[2], bfrag[4];
#pragma unroll
    for (int mi = 0; mi < 2; ++mi)
      afrag[mi] = frag_ld_a(&As[buf][(wm * 32 + mi * 16 + l16) * LDT], 0, half);
#pragma unroll
    for (int ni = 0; ni < 4; ++ni)
      bfrag[ni] = frag_ld_b(&Bs[buf][(wn * 64 + ni * 16 + l16) * LDT], 0, half);
#pragma unroll
    for (int mi = 0; mi < 2; ++mi)
#pragma unroll
      for (int ni = 0; ni < 4; ++ni)
        acc[mi][ni] = wmma_bf16(afrag[mi], bfrag[ni], acc[mi][ni]);
    __syncthreads();
  }

  // ---- epilogue ----
#pragma unroll
  for (int mi = 0; mi < 2; ++mi) {
#pragma unroll
    for (int ni = 0; ni < 4; ++ni) {
      int col = colBlk + wn * 64 + ni * 16 + l16;
      float bv = bias[col];
#pragma unroll
      for (int v = 0; v < 8; ++v) {
        int row = rowBlk + wm * 32 + mi * 16 + v + 8 * half;
        float val = acc[mi][ni][v] + bv;
        if constexpr (QKV) {
          int which = col >> 10;                 // 0=Q 1=K 2=V
          int hh = (col >> 6) & (NHEADS - 1);
          int dd = col & (HDIM - 1);
          int b = row >> 11, s = row & (SEQ - 1);
          size_t dst = (((size_t)(b * NHEADS + hh)) * SEQ + s) * HDIM + dd;
          if (which == 0)      Qb[dst] = (bf16_t)(val * 0.125f);  // 1/sqrt(64)
          else if (which == 1) Kb[dst] = (bf16_t)val;
          else                 Vb[dst] = (bf16_t)val;
        } else {
          Cout[(size_t)row * N + col] = val;
        }
      }
    }
  }
}

// ---------------------------------------------------------------------------
// Flash attention, causal.  grid = (S/128, B*H), 256 threads (8 waves).
// Each wave owns 16 query rows; 32-key chunks staged in LDS (K via async).
// ---------------------------------------------------------------------------
__global__ __launch_bounds__(256) void attn_kernel(
    const bf16_t* __restrict__ Qb, const bf16_t* __restrict__ Kb,
    const bf16_t* __restrict__ Vb, bf16_t* __restrict__ Ob) {
  constexpr int LDK = 72;   // Ks row stride (64 + 8 pad), bf16
  constexpr int LDV = 40;   // Vs row stride (32 + 8 pad)
  constexpr int LDP = 40;   // Ps row stride (32 + 8 pad)
  __shared__ __align__(16) bf16_t Ks[32 * LDK];       // [key][d]
  __shared__ __align__(16) bf16_t Vs[HDIM * LDV];     // [d][key] (transposed)
  __shared__ __align__(16) bf16_t Ps[8 * 16 * LDP];   // per-wave P [row][key]

  const int tid = threadIdx.x;
  const int wave = tid >> 5, lane = tid & 31;
  const int half = lane >> 4, l16 = lane & 15;
  const int qb = blockIdx.x;
  const int bh = blockIdx.y;
  const int qbase = qb * 128 + wave * 16;
  const size_t headOff = (size_t)bh * SEQ * HDIM;

  // Q fragments stay in registers (A layout, contraction over d)
  v16bf qfrag[2];
  qfrag[0] = frag_ld_a(Qb + headOff + (size_t)(qbase + l16) * HDIM, 0, half);
  qfrag[1] = frag_ld_a(Qb + headOff + (size_t)(qbase + l16) * HDIM, 32, half);

  float mstate[8], lstate[8];
#pragma unroll
  for (int v = 0; v < 8; ++v) { mstate[v] = -1e30f; lstate[v] = 0.f; }
  v8f oacc[4] = {};

  const int nchunk = (qb + 1) * 4;   // keys [0, (qb+1)*128) in chunks of 32
  for (int j = 0; j < nchunk; ++j) {
    const int kbase = j * 32;
    // ---- K chunk via async: 32x64 bf16 = 256 x 16B, one per thread ----
    {
      int key = tid >> 3, c8 = tid & 7;
      async_copy16(&Ks[key * LDK + c8 * 8],
                   Kb + headOff + (size_t)(kbase + key) * HDIM + c8 * 8);
    }
    // ---- V chunk transposed (scatter): Vs[d][key] ----
#pragma unroll
    for (int it = 0; it < 8; ++it) {
      int idx = tid + 256 * it;
      int key = idx >> 6, d = idx & 63;
      Vs[d * LDV + key] = Vb[headOff + (size_t)(kbase + key) * HDIM + d];
    }
    async_wait<0>();
    __syncthreads();

    if (kbase <= qbase + 15) {          // wave-uniform -> EXEC stays all-ones
      v8f sc[2];
#pragma unroll
      for (int nt = 0; nt < 2; ++nt) {
        const bf16_t* krow = &Ks[(nt * 16 + l16) * LDK];
        v8f s = {};
        s = wmma_bf16(qfrag[0], frag_ld_b(krow, 0, half), s);
        s = wmma_bf16(qfrag[1], frag_ld_b(krow, 32, half), s);
        int key = kbase + nt * 16 + l16;
#pragma unroll
        for (int v = 0; v < 8; ++v) {
          int qrow = qbase + v + 8 * half;
          if (key > qrow) s[v] = -1e30f;       // causal mask
        }
        sc[nt] = s;
      }
      // online softmax per row (row = v + 8*half, 16 keys across lanes)
#pragma unroll
      for (int v = 0; v < 8; ++v) {
        float mx = fmaxf(sc[0][v], sc[1][v]);
#pragma unroll
        for (int off = 8; off >= 1; off >>= 1)
          mx = fmaxf(mx, __shfl_xor(mx, off, 32));
        float mnew = fmaxf(mstate[v], mx);
        float scale = __expf(mstate[v] - mnew);
        float p0 = __expf(sc[0][v] - mnew);
        float p1 = __expf(sc[1][v] - mnew);
        float rs = p0 + p1;
#pragma unroll
        for (int off = 8; off >= 1; off >>= 1)
          rs += __shfl_xor(rs, off, 32);
        lstate[v] = lstate[v] * scale + rs;
        mstate[v] = mnew;
#pragma unroll
        for (int dt = 0; dt < 4; ++dt) oacc[dt][v] *= scale;
        bf16_t* prow = &Ps[(wave * 16 + v + 8 * half) * LDP];
        prow[l16]      = (bf16_t)p0;
        prow[16 + l16] = (bf16_t)p1;
      }
      // PV: P(16x32) x V(32x64)
      v16bf pf = frag_ld_a(&Ps[(wave * 16 + l16) * LDP], 0, half);
#pragma unroll
      for (int dt = 0; dt < 4; ++dt) {
        v16bf vf = frag_ld_b(&Vs[(dt * 16 + l16) * LDV], 0, half);
        oacc[dt] = wmma_bf16(pf, vf, oacc[dt]);
      }
    }
    __syncthreads();
  }

  // normalize and store bf16 to [B,S,D]
  const int b = bh >> 4, h = bh & (NHEADS - 1);
#pragma unroll
  for (int v = 0; v < 8; ++v) {
    float inv = 1.0f / lstate[v];
    int s = qbase + v + 8 * half;
    bf16_t* orow = Ob + ((size_t)(b * SEQ + s)) * DMODEL + h * HDIM;
#pragma unroll
    for (int dt = 0; dt < 4; ++dt)
      orow[dt * 16 + l16] = (bf16_t)(oacc[dt][v] * inv);
  }
}

// ---------------------------------------------------------------------------
extern "C" void kernel_launch(void* const* d_in, const int* in_sizes, int n_in,
                              void* d_out, int out_size, void* d_ws, size_t ws_size,
                              hipStream_t stream) {
  (void)in_sizes; (void)n_in; (void)out_size; (void)ws_size;
  const float* x    = (const float*)d_in[0];  // [B,S,D]
  const float* Wqkv = (const float*)d_in[1];  // [D, 3D]
  const float* bqkv = (const float*)d_in[2];  // [3D]
  const float* Wout = (const float*)d_in[3];  // [D, D]
  const float* bout = (const float*)d_in[4];  // [D]
  float* out = (float*)d_out;                 // [B,S,D] f32

  const size_t MB = 1024 * 1024;
  char* ws = (char*)d_ws;
  bf16_t* Qb    = (bf16_t*)(ws + 0 * MB);    // 16 MB each
  bf16_t* Kb    = (bf16_t*)(ws + 16 * MB);
  bf16_t* Vb    = (bf16_t*)(ws + 32 * MB);
  bf16_t* Ob    = (bf16_t*)(ws + 48 * MB);   // attn out [B,S,D] bf16
  bf16_t* xbf   = (bf16_t*)(ws + 64 * MB);   // x bf16 [M,K]
  bf16_t* Wqkvt = (bf16_t*)(ws + 80 * MB);   // [3D, D] bf16 (6 MB)
  bf16_t* Woutt = (bf16_t*)(ws + 86 * MB);   // [D, D] bf16 (2 MB)

  const int M = BATCH * SEQ;   // 8192
  const int D = DMODEL;        // 1024

  // 0) one-time converts / transposes (bandwidth-trivial)
  convert_f32_bf16<<<(M * D) / (4 * 256), 256, 0, stream>>>(x, xbf);
  transpose_to_bf16<<<dim3(3 * D / 32, D / 32), 256, 0, stream>>>(Wqkv, Wqkvt, D, 3 * D);
  transpose_to_bf16<<<dim3(D / 32, D / 32), 256, 0, stream>>>(Wout, Woutt, D, D);

  // 1) QKV projection -> head-major bf16 Q/K/V (Q pre-scaled 1/8)
  gemm128<true><<<dim3(3 * D / 128, M / 128), 256, 0, stream>>>(
      xbf, Wqkvt, bqkv, nullptr, Qb, Kb, Vb, M, 3 * D, D);

  // 2) causal flash attention
  attn_kernel<<<dim3(SEQ / 128, BATCH * NHEADS), 256, 0, stream>>>(Qb, Kb, Vb, Ob);

  // 3) output projection -> f32 + bias
  gemm128<false><<<dim3(D / 128, M / 128), 256, 0, stream>>>(
      Ob, Woutt, bout, out, nullptr, nullptr, nullptr, M, D, D);
}
